// MoE_28853590294607
// MI455X (gfx1250) — compile-verified, matrix operand order checked by reading, not run on hardware
//
#include <hip/hip_runtime.h>
#include <hip/hip_bf16.h>

typedef __bf16 bf16_t;
typedef __attribute__((ext_vector_type(16))) __bf16 v16bf;
typedef __attribute__((ext_vector_type(8)))  __bf16 v8bf;
typedef __attribute__((ext_vector_type(8)))  float  v8f;

constexpr int D     = 1024;
constexpr int BN    = 8 * 2048;   // 16384 tokens
constexpr int NEXP  = 4;
constexpr int LDSK  = D + 8;      // +16B row pad -> avoid LDS bank conflicts
constexpr int MT    = 4;          // 16-row sub-tiles per wave (M blocking)
constexpr int ROWS  = 16 * MT;    // 64 rows per block

union Frag16 { v16bf v; v8bf h[2]; };

// ---------------------------------------------------------------------------
// Gate kernel: one wave per token. Computes gate scores (D-dot x 4), top-2
// softmax weights, writes bf16(x) and initializes d_out with the residual x.
// ---------------------------------------------------------------------------
__global__ __launch_bounds__(256)
void moe_gate_kernel(const float* __restrict__ x,
                     const float* __restrict__ gW,   // [D][4]
                     const float* __restrict__ gb,   // [4]
                     bf16_t* __restrict__ xbf,
                     float* __restrict__ outres,
                     float* __restrict__ gatew)      // [BN][4]
{
    const int wave = threadIdx.x >> 5;
    const int lane = threadIdx.x & 31;
    const int t    = blockIdx.x * 8 + wave;
    const float* xr = x + (size_t)t * D;

    float s0 = 0.f, s1 = 0.f, s2 = 0.f, s3 = 0.f;
#pragma unroll 4
    for (int i = 0; i < 32; ++i) {
        const int k = lane + 32 * i;
        const float v = xr[k];
        xbf[(size_t)t * D + k]    = (bf16_t)v;
        outres[(size_t)t * D + k] = v;             // residual init
        const float4 w = *(const float4*)&gW[k * 4];
        s0 += v * w.x; s1 += v * w.y; s2 += v * w.z; s3 += v * w.w;
    }
#pragma unroll
    for (int m = 1; m < 32; m <<= 1) {
        s0 += __shfl_xor(s0, m, 32);
        s1 += __shfl_xor(s1, m, 32);
        s2 += __shfl_xor(s2, m, 32);
        s3 += __shfl_xor(s3, m, 32);
    }
    if (lane == 0) {
        float s[4] = { s0 + gb[0], s1 + gb[1], s2 + gb[2], s3 + gb[3] };
        int i0 = 0;
#pragma unroll
        for (int e = 1; e < 4; ++e) if (s[e] > s[i0]) i0 = e;
        int i1 = -1;
#pragma unroll
        for (int e = 0; e < 4; ++e) {
            if (e == i0) continue;
            if (i1 < 0 || s[e] > s[i1]) i1 = e;
        }
        const float mx = fmaxf(s[i0], s[i1]);
        const float e0 = __expf(s[i0] - mx);
        const float e1 = __expf(s[i1] - mx);
        const float inv = 1.f / (e0 + e1);
        float w4[4] = {0.f, 0.f, 0.f, 0.f};
        w4[i0] = e0 * inv;
        w4[i1] = e1 * inv;
        float4 r; r.x = w4[0]; r.y = w4[1]; r.z = w4[2]; r.w = w4[3];
        *(float4*)&gatew[(size_t)t * 4] = r;
    }
}

// ---------------------------------------------------------------------------
// Weight convert+transpose: W[e][k][n] f32 -> WT[e][n][k] bf16 (N-major so the
// WMMA B fragment loads are contiguous 16B chunks along K).
// ---------------------------------------------------------------------------
__global__ __launch_bounds__(256)
void moe_wconv_kernel(const float* __restrict__ W1, const float* __restrict__ W2,
                      bf16_t* __restrict__ WT1, bf16_t* __restrict__ WT2)
{
    __shared__ float tile[32][33];
    const int mat = blockIdx.z;  // 0..3 -> W1 experts, 4..7 -> W2 experts
    const float*  src = (mat < 4) ? (W1 + (size_t)mat * D * D)
                                  : (W2 + (size_t)(mat - 4) * D * D);
    bf16_t*       dst = (mat < 4) ? (WT1 + (size_t)mat * D * D)
                                  : (WT2 + (size_t)(mat - 4) * D * D);
    const int kb = blockIdx.y * 32;
    const int nb = blockIdx.x * 32;
    const int tx = threadIdx.x, ty = threadIdx.y;   // block (32, 8)
#pragma unroll
    for (int i = 0; i < 4; ++i)
        tile[ty + 8 * i][tx] = src[(size_t)(kb + ty + 8 * i) * D + nb + tx];
    __syncthreads();
#pragma unroll
    for (int i = 0; i < 4; ++i)
        dst[(size_t)(nb + ty + 8 * i) * D + kb + tx] = (bf16_t)tile[tx][ty + 8 * i];
}

// ---------------------------------------------------------------------------
// Fused GEMM + bias + LayerNorm (+ReLU / +gated accumulate).
// Block = 256 threads = 8 waves. Block computes ROWS(64) x 1024 outputs:
// each wave owns a 128-col slice and MT=4 16-row sub-tiles, so every B
// fragment is reused 4x (cuts B-operand L2 traffic 4x vs M=16 tiling).
// A tile (64 x 1024 bf16, ~129 KB) staged once in LDS (CDNA5: 320 KB/WG).
// ---------------------------------------------------------------------------
template<int LAYER>
__global__ __launch_bounds__(256)
void moe_gemm_ln_kernel(const bf16_t* __restrict__ A,     // [BN][D] bf16
                        const bf16_t* __restrict__ WT,    // [D(n)][D(k)] bf16
                        const float*  __restrict__ bias,  // [D]
                        const float*  __restrict__ gamma, // [D]
                        const float*  __restrict__ beta,  // [D]
                        const float*  __restrict__ gatew, // [BN][4] (layer 2)
                        bf16_t* __restrict__ Hout,        // layer-1 output
                        float*  __restrict__ Out,         // layer-2 accumulate
                        int expert)
{
    __shared__ __align__(16) bf16_t sA[ROWS * LDSK];
    __shared__ float rsum[ROWS];
    __shared__ float rsq[ROWS];

    const int wave   = threadIdx.x >> 5;
    const int lane   = threadIdx.x & 31;
    const int lane15 = lane & 15;
    const int half   = lane >> 4;          // 0: K lo-halves, 1: K hi-halves
    const int row0   = blockIdx.x * ROWS;
    const int n0w    = wave * 128;

    // Cooperative stage of the ROWS x 1024 bf16 A tile into LDS (b128 chunks).
    for (int i = threadIdx.x; i < ROWS * 128; i += 256) {
        const int r  = i >> 7;
        const int c8 = (i & 127) * 8;
        *(v8bf*)&sA[r * LDSK + c8] = *(const v8bf*)&A[(size_t)(row0 + r) * D + c8];
    }
    if (threadIdx.x < ROWS) { rsum[threadIdx.x] = 0.f; rsq[threadIdx.x] = 0.f; }
    __syncthreads();

    v8f acc[MT][8];
#pragma unroll
    for (int mt = 0; mt < MT; ++mt)
#pragma unroll
        for (int nt = 0; nt < 8; ++nt) acc[mt][nt] = (v8f)0.f;

#pragma unroll 1
    for (int k0 = 0; k0 < D; k0 += 32) {
        // A fragments (16x32, ISA 7.12.2 16-bit layout): lane15 = row,
        // K chunks {k0+8h .. +7} and {k0+16+8h .. +7}.
        Frag16 af[MT];
#pragma unroll
        for (int mt = 0; mt < MT; ++mt) {
            const bf16_t* pa = &sA[(mt * 16 + lane15) * LDSK + k0 + 8 * half];
            af[mt].h[0] = *(const v8bf*)pa;
            af[mt].h[1] = *(const v8bf*)(pa + 16);
        }
#pragma unroll
        for (int nt = 0; nt < 8; ++nt) {
            const int n = n0w + nt * 16 + lane15;
            const bf16_t* pb = &WT[(size_t)n * D + k0 + 8 * half];
            Frag16 bfm;
            bfm.h[0] = *(const v8bf*)pb;
            bfm.h[1] = *(const v8bf*)(pb + 16);
#pragma unroll
            for (int mt = 0; mt < MT; ++mt)
                acc[mt][nt] = __builtin_amdgcn_wmma_f32_16x16x32_bf16(
                    false, af[mt].v, false, bfm.v, (short)0, acc[mt][nt],
                    false, false);
        }
    }

    // Bias, then per-row LN statistics over all 1024 columns.
    // C layout: VGPR r, lanes 0-15 -> row r, lanes 16-31 -> row r+8.
#pragma unroll
    for (int nt = 0; nt < 8; ++nt) {
        const float bb = bias[n0w + nt * 16 + lane15];
#pragma unroll
        for (int mt = 0; mt < MT; ++mt)
#pragma unroll
            for (int r = 0; r < 8; ++r) acc[mt][nt][r] += bb;
    }
#pragma unroll
    for (int mt = 0; mt < MT; ++mt) {
        float ps[8], pq[8];
#pragma unroll
        for (int r = 0; r < 8; ++r) { ps[r] = 0.f; pq[r] = 0.f; }
#pragma unroll
        for (int nt = 0; nt < 8; ++nt)
#pragma unroll
            for (int r = 0; r < 8; ++r) {
                const float v = acc[mt][nt][r];
                ps[r] += v;
                pq[r] += v * v;
            }
        // Reduce across the 16 lanes of each half (rows r / r+8 separately).
#pragma unroll
        for (int m = 1; m < 16; m <<= 1) {
#pragma unroll
            for (int r = 0; r < 8; ++r) {
                ps[r] += __shfl_xor(ps[r], m, 16);
                pq[r] += __shfl_xor(pq[r], m, 16);
            }
        }
        if (lane15 == 0) {
#pragma unroll
            for (int r = 0; r < 8; ++r) {
                atomicAdd(&rsum[mt * 16 + r + 8 * half], ps[r]);
                atomicAdd(&rsq[mt * 16 + r + 8 * half],  pq[r]);
            }
        }
    }
    __syncthreads();

#pragma unroll
    for (int mt = 0; mt < MT; ++mt) {
        float meanv[8], rstdv[8], wrow[8];
#pragma unroll
        for (int r = 0; r < 8; ++r) {
            const int m = mt * 16 + r + 8 * half;
            const float mu  = rsum[m] * (1.0f / (float)D);
            const float var = rsq[m]  * (1.0f / (float)D) - mu * mu;
            meanv[r] = mu;
            rstdv[r] = rsqrtf(var + 1e-5f);
            if (LAYER == 2)
                wrow[r] = gatew[(size_t)(row0 + m) * NEXP + expert];
        }
#pragma unroll
        for (int nt = 0; nt < 8; ++nt) {
            const int col = n0w + nt * 16 + lane15;
            const float g  = gamma[col];
            const float bb = beta[col];
#pragma unroll
            for (int r = 0; r < 8; ++r) {
                const int row = row0 + mt * 16 + r + 8 * half;
                float v = (acc[mt][nt][r] - meanv[r]) * rstdv[r] * g + bb;
                if (LAYER == 1) {
                    v = fmaxf(v, 0.f);
                    Hout[(size_t)row * D + col] = (bf16_t)v;
                } else {
                    Out[(size_t)row * D + col] += wrow[r] * v;
                }
            }
        }
    }
}

// ---------------------------------------------------------------------------
extern "C" void kernel_launch(void* const* d_in, const int* in_sizes, int n_in,
                              void* d_out, int out_size, void* d_ws, size_t ws_size,
                              hipStream_t stream)
{
    (void)in_sizes; (void)n_in; (void)out_size; (void)ws_size;

    const float* x      = (const float*)d_in[0];   // [8,2048,1024]
    const float* gate_W = (const float*)d_in[1];   // [1024,4]
    const float* gate_b = (const float*)d_in[2];   // [4]
    const float* W1     = (const float*)d_in[3];   // [4,1024,1024]
    const float* b1     = (const float*)d_in[4];   // [4,1024]
    const float* g1     = (const float*)d_in[5];
    const float* be1    = (const float*)d_in[6];
    const float* W2     = (const float*)d_in[7];
    const float* b2     = (const float*)d_in[8];
    const float* g2     = (const float*)d_in[9];
    const float* be2    = (const float*)d_in[10];
    float* out = (float*)d_out;

    // Workspace layout (~84 MB)
    char* ws = (char*)d_ws;
    bf16_t* xbf = (bf16_t*)ws;               ws += (size_t)BN * D * sizeof(bf16_t);
    bf16_t* Hbf = (bf16_t*)ws;               ws += (size_t)BN * D * sizeof(bf16_t);
    bf16_t* WT1 = (bf16_t*)ws;               ws += (size_t)NEXP * D * D * sizeof(bf16_t);
    bf16_t* WT2 = (bf16_t*)ws;               ws += (size_t)NEXP * D * D * sizeof(bf16_t);
    float*  gatew = (float*)ws;              ws += (size_t)BN * NEXP * sizeof(float);

    // 1) Gating + residual init + bf16 conversion of x.
    moe_gate_kernel<<<BN / 8, 256, 0, stream>>>(x, gate_W, gate_b, xbf, out, gatew);

    // 2) Transpose + convert all expert weights to bf16 (N-major).
    moe_wconv_kernel<<<dim3(D / 32, D / 32, 2 * NEXP), dim3(32, 8), 0, stream>>>(
        W1, W2, WT1, WT2);

    // 3) Per expert: fused GEMM+LN+ReLU, then fused GEMM+LN + gated accumulate.
    for (int e = 0; e < NEXP; ++e) {
        moe_gemm_ln_kernel<1><<<BN / ROWS, 256, 0, stream>>>(
            xbf, WT1 + (size_t)e * D * D,
            b1 + (size_t)e * D, g1 + (size_t)e * D, be1 + (size_t)e * D,
            nullptr, Hbf, nullptr, e);
        moe_gemm_ln_kernel<2><<<BN / ROWS, 256, 0, stream>>>(
            Hbf, WT2 + (size_t)e * D * D,
            b2 + (size_t)e * D, g2 + (size_t)e * D, be2 + (size_t)e * D,
            gatew, nullptr, out, e);
    }
}
